// MHABlock_16492674416925
// MI455X (gfx1250) — matrix operand order repaired and checked by reading, not compiled, not run on hardware
//
#include <hip/hip_runtime.h>
#include <stdint.h>

typedef __attribute__((ext_vector_type(16))) __bf16 v16bf;
typedef __attribute__((ext_vector_type(8)))  float  v8f;
typedef __attribute__((ext_vector_type(4)))  unsigned short ush4;
typedef __attribute__((ext_vector_type(8)))  unsigned short ush8;
typedef __attribute__((ext_vector_type(16))) unsigned short ush16;

static constexpr int cB = 8, cN = 2048, cC = 1024, cH = 16, cD = 64;
static constexpr int ROWS = cB * cN;     // 16384
static constexpr int NQKV = 3 * cC;      // 3072

__device__ inline unsigned short f2bf(float f) {
  unsigned u = __builtin_bit_cast(unsigned, f);
  u += 0x7fffu + ((u >> 16) & 1u);       // round-to-nearest-even
  return (unsigned short)(u >> 16);
}
__device__ inline float bf2f(unsigned short h) {
  unsigned u = ((unsigned)h) << 16;
  return __builtin_bit_cast(float, u);
}
union BFPack { ush16 u; v16bf b; };
__device__ inline v16bf pack16(ush8 lo, ush8 hi) {
  BFPack p;
#pragma unroll
  for (int i = 0; i < 8; ++i) { p.u[i] = lo[i]; p.u[8 + i] = hi[i]; }
  return p.b;
}
__device__ inline v8f zero8() {
  v8f z;
#pragma unroll
  for (int i = 0; i < 8; ++i) z[i] = 0.0f;
  return z;
}
// Issue one 16-byte async global->LDS copy (ASYNCcnt-tracked, no VGPR data path).
__device__ inline void async_b128(const void* gptr, void* lptr) {
  unsigned loff = (unsigned)(uintptr_t)lptr;                 // LDS offset (low 32 bits)
  unsigned long long ga = (unsigned long long)(uintptr_t)gptr;
  asm volatile("global_load_async_to_lds_b128 %0, %1, off"
               :: "v"(loff), "v"(ga) : "memory");
}

// ---------------- BatchNorm statistics (float4 per thread) ----------------
__global__ __launch_bounds__(256) void k_bn_stats(const float* __restrict__ x,
                                                  float* __restrict__ chsum,
                                                  float* __restrict__ chss) {
  int t = threadIdx.x;
  int c0 = t * 4;
  int row0 = blockIdx.x * 64;
  float s[4] = {0.f, 0.f, 0.f, 0.f}, q[4] = {0.f, 0.f, 0.f, 0.f};
  for (int r = 0; r < 64; ++r) {
    float4 v = *(const float4*)(x + (size_t)(row0 + r) * cC + c0);
    s[0] += v.x; q[0] += v.x * v.x;
    s[1] += v.y; q[1] += v.y * v.y;
    s[2] += v.z; q[2] += v.z * v.z;
    s[3] += v.w; q[3] += v.w * v.w;
  }
#pragma unroll
  for (int j = 0; j < 4; ++j) {
    atomicAdd(&chsum[c0 + j], s[j]);
    atomicAdd(&chss[c0 + j], q[j]);
  }
}

__global__ __launch_bounds__(256) void k_bn_fin(const float* __restrict__ chsum,
                                                const float* __restrict__ chss,
                                                const float* __restrict__ gamma,
                                                const float* __restrict__ beta,
                                                float* __restrict__ scaleC,
                                                float* __restrict__ biasC) {
  int c = blockIdx.x * 256 + threadIdx.x;
  if (c >= cC) return;
  float inv_n = 1.0f / (float)ROWS;
  float mean = chsum[c] * inv_n;
  float var = chss[c] * inv_n - mean * mean;
  float iv = rsqrtf(var + 1e-5f);
  float sc = gamma[c] * iv;
  scaleC[c] = sc;
  biasC[c] = beta[c] - mean * sc;
}

// ---------------- Apply BN, write bf16 normalized activations ----------------
__global__ __launch_bounds__(256) void k_bn_apply(const float* __restrict__ hs,
                                                  const float* __restrict__ scaleC,
                                                  const float* __restrict__ biasC,
                                                  unsigned short* __restrict__ xnbf) {
  size_t idx = ((size_t)blockIdx.x * 256 + threadIdx.x) * 4;
  int c = (int)(idx & (cC - 1));
  float4 v = *(const float4*)(hs + idx);
  float4 sc = *(const float4*)(scaleC + c);
  float4 bi = *(const float4*)(biasC + c);
  ush4 o;
  o[0] = f2bf(v.x * sc.x + bi.x);
  o[1] = f2bf(v.y * sc.y + bi.y);
  o[2] = f2bf(v.z * sc.z + bi.z);
  o[3] = f2bf(v.w * sc.w + bi.w);
  *(ush4*)(xnbf + idx) = o;
}

// ---------------- Transpose + convert weights: W (K x Nc fp32) -> Wt (Nc x K bf16) ----------------
__global__ __launch_bounds__(256) void k_wt(const float* __restrict__ W, int Kdim, int Nc,
                                            unsigned short* __restrict__ Wt) {
  __shared__ unsigned short tile[64][72];   // pad rows to 144B (8B-aligned chunks)
  int t = threadIdx.x;
  int n0 = blockIdx.x * 64, k0 = blockIdx.y * 64;
#pragma unroll
  for (int it = 0; it < 16; ++it) {
    int e = t + 256 * it;
    int kk = e >> 6, nn = e & 63;
    tile[nn][kk] = f2bf(W[(size_t)(k0 + kk) * Nc + n0 + nn]);
  }
  __syncthreads();
#pragma unroll
  for (int it = 0; it < 4; ++it) {
    int ch = t + 256 * it;              // 1024 chunks of 4
    int nn = ch >> 4, kq = (ch & 15) * 4;
    *(ush4*)&Wt[(size_t)(n0 + nn) * Kdim + k0 + kq] = *(const ush4*)&tile[nn][kq];
  }
}

// ---------------- bf16 WMMA GEMM with async double-buffered LDS staging ----------------
// A: M x K bf16 (lda). Wt: Nc x K bf16 (ldw, K-contiguous). Out: M x Nc fp32 (ldc).
__global__ __launch_bounds__(256) void k_gemm_bf(const unsigned short* __restrict__ A, int lda,
                                                 const unsigned short* __restrict__ Wt, int ldw,
                                                 const float* __restrict__ bias,
                                                 float* __restrict__ Out, int ldc, int K) {
  __shared__ unsigned short lA[2][128][32];   // [buf][m][k]
  __shared__ unsigned short lB[2][128][32];   // [buf][n][k]
  int t = threadIdx.x, lane = t & 31, w = t >> 5;
  int half = lane >> 4, l16 = lane & 15;
  int wm = w & 3, wn = w >> 2;                // 4x2 wave grid, each wave 32x64
  int m0 = blockIdx.y * 128, n0 = blockIdx.x * 128;
  v8f acc[2][4];
#pragma unroll
  for (int i = 0; i < 2; ++i)
#pragma unroll
    for (int j = 0; j < 4; ++j) acc[i][j] = zero8();

  // per-thread staging coordinates (2 x 16B chunks per operand per chunk)
  auto issue = [&](int k0, int s) {
#pragma unroll
    for (int it = 0; it < 2; ++it) {
      int ch = t + 256 * it;                 // 512 chunks of 8 bf16 (16B)
      int r = ch >> 2, kq = (ch & 3) * 8;
      async_b128(&A[(size_t)(m0 + r) * lda + k0 + kq], &lA[s][r][kq]);
      async_b128(&Wt[(size_t)(n0 + r) * ldw + k0 + kq], &lB[s][r][kq]);
    }
  };

  issue(0, 0);                               // prologue: stage chunk 0
  int sel = 0;
  for (int k0 = 0; k0 < K; k0 += 32) {
    if (k0 + 32 < K) {
      issue(k0 + 32, sel ^ 1);               // overlap next copy with this compute
      asm volatile("s_wait_asynccnt 0x4" ::: "memory");   // older 4 ops (this chunk) done
    } else {
      asm volatile("s_wait_asynccnt 0x0" ::: "memory");
    }
    __syncthreads();                         // all waves' LDS writes visible

    v16bf afr[2], bfr[4];
#pragma unroll
    for (int tr = 0; tr < 2; ++tr) {
      int rm = wm * 32 + tr * 16 + l16;
      afr[tr] = pack16(*(const ush8*)&lA[sel][rm][8 * half],
                       *(const ush8*)&lA[sel][rm][16 + 8 * half]);
    }
#pragma unroll
    for (int tc = 0; tc < 4; ++tc) {
      int cn = wn * 64 + tc * 16 + l16;
      bfr[tc] = pack16(*(const ush8*)&lB[sel][cn][8 * half],
                       *(const ush8*)&lB[sel][cn][16 + 8 * half]);
    }
#pragma unroll
    for (int tr = 0; tr < 2; ++tr)
#pragma unroll
      for (int tc = 0; tc < 4; ++tc)
        acc[tr][tc] = __builtin_amdgcn_wmma_f32_16x16x32_bf16(
            false, afr[tr], false, bfr[tc], (short)0, acc[tr][tc], false, false);
    __syncthreads();                         // reads done before buffer reuse
    sel ^= 1;
  }
#pragma unroll
  for (int tr = 0; tr < 2; ++tr)
#pragma unroll
    for (int tc = 0; tc < 4; ++tc) {
      int col = n0 + wn * 64 + tc * 16 + l16;
      float bv = bias[col];
#pragma unroll
      for (int r = 0; r < 8; ++r) {
        int row = m0 + wm * 32 + tr * 16 + r + 8 * half;
        Out[(size_t)row * ldc + col] = acc[tr][tc][r] + bv;
      }
    }
}

// ---------------- Focused feature map (q and k), writes bf16 ----------------
__global__ __launch_bounds__(256) void k_focus(const float* __restrict__ qkv,
                                               const float* __restrict__ scale_p,
                                               unsigned short* __restrict__ qf,
                                               unsigned short* __restrict__ kf) {
  __shared__ float red[256];
  int t = threadIdx.x;
  int row = blockIdx.x >> 1;
  int which = blockIdx.x & 1;
  const float* src = qkv + (size_t)row * NQKV + which * cC;
  unsigned short* dst = (which ? kf : qf) + (size_t)row * cC;
  int c0 = t * 4;
  float4 v4 = *(const float4*)(src + c0);
  float4 sp4 = *(const float4*)(scale_p + c0);
  float x[4];
  x[0] = v4.x; x[1] = v4.y; x[2] = v4.z; x[3] = v4.w;
  float sp[4];
  sp[0] = sp4.x; sp[1] = sp4.y; sp[2] = sp4.z; sp[3] = sp4.w;
  float ss = 0.f;
#pragma unroll
  for (int j = 0; j < 4; ++j) {
    float v = (x[j] > 0.f ? x[j] : 0.f) + 1e-6f;
    v = v / log1pf(expf(sp[j]));           // softplus scale
    x[j] = v;
    ss += v * v;
  }
  red[t] = ss; __syncthreads();
  for (int s = 128; s > 0; s >>= 1) { if (t < s) red[t] += red[t + s]; __syncthreads(); }
  float s1 = red[0]; __syncthreads();
  float x3[4];
  float ss2 = 0.f;
#pragma unroll
  for (int j = 0; j < 4; ++j) { float c3 = x[j] * x[j] * x[j]; x3[j] = c3; ss2 += c3 * c3; }
  red[t] = ss2; __syncthreads();
  for (int s = 128; s > 0; s >>= 1) { if (t < s) red[t] += red[t + s]; __syncthreads(); }
  float s2 = red[0];
  float factor = sqrtf(s1) * rsqrtf(s2);   // restore original norm
  ush4 o;
#pragma unroll
  for (int j = 0; j < 4; ++j) o[j] = f2bf(x3[j] * factor);
  *(ush4*)(dst + c0) = o;
}

// ---------------- k-sum over sequence per (b,h) ----------------
__global__ __launch_bounds__(256) void k_ksum(const unsigned short* __restrict__ kf,
                                              float* __restrict__ ksum) {
  __shared__ float red[256];
  int t = threadIdx.x;
  int bh = blockIdx.x; int b = bh / cH, h = bh % cH;
  int d = t & 63, seg = t >> 6;
  float s = 0.f;
  for (int n = seg; n < cN; n += 4)
    s += bf2f(kf[(size_t)(b * cN + n) * cC + h * cD + d]);
  red[t] = s; __syncthreads();
  if (t < 64)
    ksum[(size_t)b * cC + h * cD + t] = red[t] + red[t + 64] + red[t + 128] + red[t + 192];
}

// ---------------- kv = k^T v (64x64 per head) via WMMA ----------------
__global__ __launch_bounds__(256) void k_kv(const unsigned short* __restrict__ kf,
                                            const float* __restrict__ qkv,
                                            float* __restrict__ kvbuf) {
  __shared__ unsigned short lK[64][32];    // [d][n]
  __shared__ unsigned short lV[64][32];    // [e][n]
  int t = threadIdx.x, lane = t & 31, w = t >> 5;
  int half = lane >> 4, l16 = lane & 15;
  int wm = w & 3, wn = w >> 2;
  int bh = blockIdx.x; int b = bh / cH, h = bh % cH;
  v8f acc[2]; acc[0] = zero8(); acc[1] = zero8();
  for (int n0 = 0; n0 < cN; n0 += 32) {
#pragma unroll
    for (int it = 0; it < 2; ++it) {
      int ch = t + 256 * it;             // 512 chunks of 4 elements
      int n = ch >> 4, dq = (ch & 15) * 4;
      size_t rbase = (size_t)(b * cN + n0 + n);
      ush4 k4 = *(const ush4*)&kf[rbase * cC + h * cD + dq];
      float4 v4 = *(const float4*)&qkv[rbase * NQKV + 2 * cC + h * cD + dq];
      lK[dq + 0][n] = k4[0]; lK[dq + 1][n] = k4[1];
      lK[dq + 2][n] = k4[2]; lK[dq + 3][n] = k4[3];
      lV[dq + 0][n] = f2bf(v4.x); lV[dq + 1][n] = f2bf(v4.y);
      lV[dq + 2][n] = f2bf(v4.z); lV[dq + 3][n] = f2bf(v4.w);
    }
    __syncthreads();
    int rm = wm * 16 + l16;
    v16bf a = pack16(*(const ush8*)&lK[rm][8 * half],
                     *(const ush8*)&lK[rm][16 + 8 * half]);
#pragma unroll
    for (int tc = 0; tc < 2; ++tc) {
      int cn = wn * 32 + tc * 16 + l16;
      v16bf bb = pack16(*(const ush8*)&lV[cn][8 * half],
                        *(const ush8*)&lV[cn][16 + 8 * half]);
      acc[tc] = __builtin_amdgcn_wmma_f32_16x16x32_bf16(
          false, a, false, bb, (short)0, acc[tc], false, false);
    }
    __syncthreads();
  }
#pragma unroll
  for (int tc = 0; tc < 2; ++tc) {
    int e = wn * 32 + tc * 16 + l16;
#pragma unroll
    for (int r = 0; r < 8; ++r) {
      int d = wm * 16 + r + 8 * half;
      kvbuf[((size_t)bh * cD + d) * cD + e] = acc[tc][r];
    }
  }
}

// ---------------- z = 1/(q . ksum + 1e-6) per (b,n,h) ----------------
__global__ __launch_bounds__(256) void k_z(const unsigned short* __restrict__ qf,
                                           const float* __restrict__ ksum,
                                           float* __restrict__ z) {
  __shared__ float prod[1024];
  int t = threadIdx.x;
  int row = blockIdx.x;                    // b*N + n
  int b = row / cN;
  int c0 = t * 4;
  ush4 q4 = *(const ush4*)&qf[(size_t)row * cC + c0];
  float4 k4 = *(const float4*)&ksum[(size_t)b * cC + c0];
  prod[c0 + 0] = bf2f(q4[0]) * k4.x;
  prod[c0 + 1] = bf2f(q4[1]) * k4.y;
  prod[c0 + 2] = bf2f(q4[2]) * k4.z;
  prod[c0 + 3] = bf2f(q4[3]) * k4.w;
  __syncthreads();
  if (t < cH) {
    float s = 0.f;
    for (int d = 0; d < cD; ++d) s += prod[t * cD + d];
    z[(size_t)row * cH + t] = 1.0f / (s + 1e-6f);
  }
}

// ---------------- o = (q @ kv) * z via WMMA ----------------
__global__ __launch_bounds__(256) void k_o(const unsigned short* __restrict__ qf,
                                           const float* __restrict__ kvbuf,
                                           const float* __restrict__ z,
                                           float* __restrict__ obuf) {
  __shared__ unsigned short lQ[128][32];   // [n][k]
  __shared__ unsigned short lKV[64][32];   // [e][k]
  int t = threadIdx.x, lane = t & 31, w = t >> 5;
  int half = lane >> 4, l16 = lane & 15;
  int bh = blockIdx.y; int b = bh / cH, h = bh % cH;
  int n0 = blockIdx.x * 128;
  v8f acc[4];
#pragma unroll
  for (int i = 0; i < 4; ++i) acc[i] = zero8();
  for (int kc = 0; kc < cD; kc += 32) {
#pragma unroll
    for (int it = 0; it < 2; ++it) {
      int ch = t + 256 * it;             // 512 chunks of 8 bf16
      int r = ch >> 2, kq = (ch & 3) * 8;
      *(ush8*)&lQ[r][kq] =
          *(const ush8*)&qf[(size_t)(b * cN + n0 + r) * cC + h * cD + kc + kq];
    }
#pragma unroll
    for (int it = 0; it < 2; ++it) {
      int ch = t + 256 * it;             // 512 chunks of 4 floats
      int kk = ch >> 4, eq = (ch & 15) * 4;
      float4 v4 = *(const float4*)&kvbuf[((size_t)bh * cD + kc + kk) * cD + eq];
      lKV[eq + 0][kk] = f2bf(v4.x); lKV[eq + 1][kk] = f2bf(v4.y);
      lKV[eq + 2][kk] = f2bf(v4.z); lKV[eq + 3][kk] = f2bf(v4.w);
    }
    __syncthreads();
    int rm = w * 16 + l16;
    v16bf a = pack16(*(const ush8*)&lQ[rm][8 * half],
                     *(const ush8*)&lQ[rm][16 + 8 * half]);
#pragma unroll
    for (int tc = 0; tc < 4; ++tc) {
      int cn = tc * 16 + l16;
      v16bf bb = pack16(*(const ush8*)&lKV[cn][8 * half],
                        *(const ush8*)&lKV[cn][16 + 8 * half]);
      acc[tc] = __builtin_amdgcn_wmma_f32_16x16x32_bf16(
          false, a, false, bb, (short)0, acc[tc], false, false);
    }
    __syncthreads();
  }
#pragma unroll
  for (int tc = 0; tc < 4; ++tc) {
    int col = h * cD + tc * 16 + l16;
#pragma unroll
    for (int r = 0; r < 8; ++r) {
      int n = n0 + w * 16 + r + 8 * half;
      float zz = z[(size_t)(b * cN + n) * cH + h];
      obuf[(size_t)(b * cN + n) * cC + col] = acc[tc][r] * zz;
    }
  }
}

// ---------------- depthwise conv residual on v, t = o + dwc -> bf16 ----------------
__global__ __launch_bounds__(256) void k_dwc(const float* __restrict__ qkv,
                                             const float* __restrict__ dw,
                                             const float* __restrict__ db,
                                             const float* __restrict__ obuf,
                                             unsigned short* __restrict__ tbf) {
  size_t idx = ((size_t)blockIdx.x * 256 + threadIdx.x) * 4;
  int c = (int)(idx & (cC - 1));
  int rn = (int)(idx >> 10);               // global row b*N+n
  int n = rn & (cN - 1);
  float4 acc = *(const float4*)(obuf + idx);
  acc.x += db[c]; acc.y += db[c + 1]; acc.z += db[c + 2]; acc.w += db[c + 3];
#pragma unroll
  for (int j = 0; j < 5; ++j) {
    int nn = n + j - 2;
    if (nn >= 0 && nn < cN) {
      float4 v4 = *(const float4*)(qkv + (size_t)(rn + j - 2) * NQKV + 2 * cC + c);
      acc.x += v4.x * dw[(c + 0) * 5 + j];
      acc.y += v4.y * dw[(c + 1) * 5 + j];
      acc.z += v4.z * dw[(c + 2) * 5 + j];
      acc.w += v4.w * dw[(c + 3) * 5 + j];
    }
  }
  ush4 o;
  o[0] = f2bf(acc.x); o[1] = f2bf(acc.y); o[2] = f2bf(acc.z); o[3] = f2bf(acc.w);
  *(ush4*)(tbf + idx) = o;
}

extern "C" void kernel_launch(void* const* d_in, const int* in_sizes, int n_in,
                              void* d_out, int out_size, void* d_ws, size_t ws_size,
                              hipStream_t stream) {
  (void)in_sizes; (void)n_in; (void)out_size; (void)ws_size;
  const float* hs     = (const float*)d_in[0];
  const float* gamma  = (const float*)d_in[1];
  const float* beta   = (const float*)d_in[2];
  const float* Wqkv   = (const float*)d_in[3];
  const float* bqkv   = (const float*)d_in[4];
  const float* scalep = (const float*)d_in[5];
  const float* dwcw   = (const float*)d_in[6];
  const float* dwcb   = (const float*)d_in[7];
  const float* Wproj  = (const float*)d_in[8];
  const float* bproj  = (const float*)d_in[9];
  float* out = (float*)d_out;

  char* ws = (char*)d_ws;
  size_t off = 0;
  auto alloc = [&](size_t bytes) -> void* {
    void* p = ws + off;
    off = (off + bytes + 255) & ~(size_t)255;
    return p;
  };
  float*          qkv    = (float*)alloc((size_t)ROWS * NQKV * 4);          // 201 MB
  unsigned short* xnbf   = (unsigned short*)alloc((size_t)ROWS * cC * 2);   // 32 MB
  unsigned short* qf     = (unsigned short*)alloc((size_t)ROWS * cC * 2);   // 32 MB
  unsigned short* kf     = (unsigned short*)alloc((size_t)ROWS * cC * 2);   // 32 MB
  unsigned short* tbf    = (unsigned short*)alloc((size_t)ROWS * cC * 2);   // 32 MB
  unsigned short* wqkvt  = (unsigned short*)alloc((size_t)NQKV * cC * 2);   // 6 MB
  unsigned short* wprojt = (unsigned short*)alloc((size_t)cC * cC * 2);     // 2 MB
  float*          kvbuf  = (float*)alloc((size_t)cB * cH * cD * cD * 4);
  float*          ksum   = (float*)alloc((size_t)cB * cC * 4);
  float*          zbuf   = (float*)alloc((size_t)ROWS * cH * 4);
  float*          obuf   = (float*)alloc((size_t)ROWS * cC * 4);            // 64 MB
  float*          chsum  = (float*)alloc(cC * 4);
  float*          chss   = (float*)alloc(cC * 4);
  float*          scaleC = (float*)alloc(cC * 4);
  float*          biasC  = (float*)alloc(cC * 4);

  hipMemsetAsync(chsum, 0, 2 * cC * 4, stream);   // chsum + chss contiguous

  k_bn_stats<<<256, 256, 0, stream>>>(hs, chsum, chss);
  k_bn_fin<<<4, 256, 0, stream>>>(chsum, chss, gamma, beta, scaleC, biasC);
  k_bn_apply<<<ROWS * cC / 1024, 256, 0, stream>>>(hs, scaleC, biasC, xnbf);

  // one-time weight transpose+convert (per launch)
  k_wt<<<dim3(NQKV / 64, cC / 64), 256, 0, stream>>>(Wqkv, cC, NQKV, wqkvt);
  k_wt<<<dim3(cC / 64, cC / 64), 256, 0, stream>>>(Wproj, cC, cC, wprojt);

  // QKV GEMM (bf16 operands, fp32 accumulate, async double-buffered staging)
  k_gemm_bf<<<dim3(NQKV / 128, ROWS / 128), 256, 0, stream>>>(
      xnbf, cC, wqkvt, cC, bqkv, qkv, NQKV, cC);

  k_focus<<<2 * ROWS, 256, 0, stream>>>(qkv, scalep, qf, kf);
  k_ksum<<<cB * cH, 256, 0, stream>>>(kf, ksum);
  k_kv<<<cB * cH, 256, 0, stream>>>(kf, qkv, kvbuf);
  k_z<<<ROWS, 256, 0, stream>>>(qf, ksum, zbuf);
  k_o<<<dim3(cN / 128, cB * cH), 256, 0, stream>>>(qf, kvbuf, zbuf, obuf);
  k_dwc<<<ROWS * cC / 1024, 256, 0, stream>>>(qkv, dwcw, dwcb, obuf, tbf);

  // Final projection
  k_gemm_bf<<<dim3(cC / 128, ROWS / 128), 256, 0, stream>>>(
      tbf, cC, wprojt, cC, bproj, out, cC, cC);
}
